// LightXML_54116587930095
// MI455X (gfx1250) — matrix operand order repaired and checked by reading, compiled, not run on hardware
//
#include <hip/hip_runtime.h>
#include <hip/hip_bf16.h>
#include <math.h>

// ---------------------------------------------------------------------------
// LightXML forward for MI455X (gfx1250, wave32, WMMA)
//   out = (group_logits [256,8192] f32,
//          candidates   [256,820]  i32,
//          cand_scores  [256,820]  f32)
// One-shot f32->bf16 conversion of X/W0/W1 into workspace, then a
// conversion-free WMMA GEMM: B fragments via global_load_b128 (bf16 W0 is
// 63 MB -> L2-resident in 192 MB), A tile staged to LDS with async
// global->LDS copies (ASYNCcnt) and read back via ds_load_b128.
// ---------------------------------------------------------------------------

#define BATCH 256
#define FDIM  3840
#define NGRP  8192
#define SLAB  82
#define TOPK  10
#define HDIM  300
#define NCAND (TOPK * SLAB)   // 820
#define KC    384             // K chunk staged in LDS (3840 = 10*384)
#define LDW   (FDIM / 2)      // bf16 row length in dwords (1920)

// Async global->LDS staging (inline asm, §15.18.3 / CDNA5 ISA).
// Flip to 0 to fall back to the synchronous VGPR round-trip copy.
#define USE_ASYNC_LDS 1

typedef __attribute__((ext_vector_type(16))) __bf16    v16bf;
typedef __attribute__((ext_vector_type(8)))  float     v8f;
typedef __attribute__((ext_vector_type(8)))  unsigned  v8u;

// pack two f32 -> one dword of 2 bf16 (round-to-nearest-even)
__device__ __forceinline__ unsigned pk_bf16(float lo, float hi) {
#if __has_builtin(__builtin_amdgcn_cvt_pk_bf16_f32)
    typedef __attribute__((ext_vector_type(2))) __bf16 v2bf_t;
    v2bf_t r = __builtin_amdgcn_cvt_pk_bf16_f32(lo, hi);
    return __builtin_bit_cast(unsigned, r);
#else
    unsigned a = __builtin_bit_cast(unsigned, lo);
    unsigned b = __builtin_bit_cast(unsigned, hi);
    a += 0x7FFFu + ((a >> 16) & 1u);
    b += 0x7FFFu + ((b >> 16) & 1u);
    // result bytes [0,1]=a[2,3], [2,3]=b[2,3]  (v_perm_b32: src1 bytes 0-3)
    return __builtin_amdgcn_perm(b, a, 0x07060302u);
#endif
}

__device__ __forceinline__ float sigmoidf_(float x) {
    return 1.0f / (1.0f + __expf(-x));
}

__device__ __forceinline__ void wait_asynccnt0() {
#if __has_builtin(__builtin_amdgcn_s_wait_asynccnt)
    __builtin_amdgcn_s_wait_asynccnt(0);
#else
    asm volatile("s_wait_asynccnt 0x0" ::: "memory");
#endif
}

// ---------------------------------------------------------------------------
// Streaming f32 -> packed bf16 converter (n4 = number of float4 elements)
// ---------------------------------------------------------------------------
__global__ void __launch_bounds__(256)
k_cvt_bf16(const float* __restrict__ src, unsigned* __restrict__ dst, int n4)
{
    int i = blockIdx.x * 256 + threadIdx.x;
    const int stride = gridDim.x * 256;
    for (; i < n4; i += stride) {
        float4 v = ((const float4*)src)[i];
        uint2 p;
        p.x = pk_bf16(v.x, v.y);
        p.y = pk_bf16(v.z, v.w);
        ((uint2*)dst)[i] = p;
    }
}

// ---------------------------------------------------------------------------
// Kernel 1 (fast path): group_logits = X @ W0^T + b0, all-bf16 operands.
// Block: 256 thr (8 waves). Tile M=32 x N=512; wave 32x64 (2x4 accumulators).
// Fragment layouts used here (contiguous k, same order for A and B):
//   A 16x32 bf16: lanes0-15 M, K=0..15 ; lanes16-31 M, K=16..31
//   B 32x16 bf16: lanes0-15 N, K=0..15 ; lanes16-31 N, K=16..31
// ---------------------------------------------------------------------------
__global__ void __launch_bounds__(256)
k_group_gemm_bf(const unsigned* __restrict__ Xb, const unsigned* __restrict__ W0b,
                const float* __restrict__ b0, float* __restrict__ GL)
{
    __shared__ unsigned sA[32 * (KC / 2)];   // 32 rows x 192 dwords = 24 KB

    const int tid   = threadIdx.x;
    const int wave  = tid >> 5;
    const int lane  = tid & 31;
    const int lh    = lane & 15;
    const bool hi   = (lane >= 16);
    const int rbase = blockIdx.y * 32;
    const int nbase = blockIdx.x * 512 + wave * 64;

    v8f acc[2][4];
    #pragma unroll
    for (int m = 0; m < 2; ++m)
        #pragma unroll
        for (int a = 0; a < 4; ++a)
            #pragma unroll
            for (int r = 0; r < 8; ++r) acc[m][a][r] = 0.0f;

    for (int k0 = 0; k0 < FDIM; k0 += KC) {
        __syncthreads();   // previous chunk fully consumed before overwrite
        // ---- stage A tile [32 x KC] bf16 into LDS -------------------------
        {
            const int r  = tid >> 3;                 // 0..31
            const int c8 = tid & 7;                  // 8 threads/row
            const unsigned* src = Xb + (size_t)(rbase + r) * LDW + (k0 >> 1);
            unsigned* dstRow = &sA[r * (KC / 2)];
#if USE_ASYNC_LDS
            #pragma unroll
            for (int it = 0; it < (KC / 2) / 32; ++it) {     // 6 x b128 each
                const int c = (c8 + it * 8) * 4;             // dword index
                const unsigned ldsAddr =
                    (unsigned)(uintptr_t)(dstRow + c);       // low 32b = LDS
                const unsigned long long ga =
                    (unsigned long long)(uintptr_t)(src + c);
                asm volatile(
                    "global_load_async_to_lds_b128 %0, %1, off"
                    :: "v"(ldsAddr), "v"(ga) : "memory");
            }
            wait_asynccnt0();
#else
            const uint4* s4 = (const uint4*)src;
            uint4* d4 = (uint4*)dstRow;
            #pragma unroll
            for (int it = 0; it < (KC / 2) / 32; ++it)
                d4[c8 + it * 8] = s4[c8 + it * 8];
#endif
        }
        __syncthreads();

        for (int k = 0; k < KC; k += 32) {
            // A fragments (two ds_load_b128 each)
            v8u afu[2];
            const int dbase = (k >> 1) + (hi ? 8 : 0);
            #pragma unroll
            for (int m = 0; m < 2; ++m) {
                const unsigned* p = &sA[(m * 16 + lh) * (KC / 2) + dbase];
                uint4 lo4 = *(const uint4*)(p);
                uint4 hi4 = *(const uint4*)(p + 4);
                afu[m][0] = lo4.x; afu[m][1] = lo4.y;
                afu[m][2] = lo4.z; afu[m][3] = lo4.w;
                afu[m][4] = hi4.x; afu[m][5] = hi4.y;
                afu[m][6] = hi4.z; afu[m][7] = hi4.w;
            }
            const v16bf afrag0 = __builtin_bit_cast(v16bf, afu[0]);
            const v16bf afrag1 = __builtin_bit_cast(v16bf, afu[1]);

            #pragma unroll
            for (int a = 0; a < 4; ++a) {
                const int col = nbase + a * 16 + lh;
                const unsigned* bp =
                    W0b + (size_t)col * LDW + ((k0 + k) >> 1) + (hi ? 8 : 0);
                __builtin_prefetch(bp + (KC >> 1), 0, 0);   // next K chunk
                uint4 u0 = ((const uint4*)bp)[0];
                uint4 u1 = ((const uint4*)bp)[1];
                v8u bu;
                bu[0] = u0.x; bu[1] = u0.y; bu[2] = u0.z; bu[3] = u0.w;
                bu[4] = u1.x; bu[5] = u1.y; bu[6] = u1.z; bu[7] = u1.w;
                const v16bf bfrag = __builtin_bit_cast(v16bf, bu);

                acc[0][a] = __builtin_amdgcn_wmma_f32_16x16x32_bf16(
                    false, afrag0, false, bfrag, (short)0, acc[0][a], false, false);
                acc[1][a] = __builtin_amdgcn_wmma_f32_16x16x32_bf16(
                    false, afrag1, false, bfrag, (short)0, acc[1][a], false, false);
            }
        }
    }

    #pragma unroll
    for (int a = 0; a < 4; ++a) {
        const int col  = nbase + a * 16 + lh;
        const float bias = b0[col];
        #pragma unroll
        for (int m = 0; m < 2; ++m) {
            const int row0 = rbase + m * 16 + (hi ? 8 : 0);
            float* out = GL + (size_t)row0 * NGRP + col;
            #pragma unroll
            for (int r = 0; r < 8; ++r)
                out[(size_t)r * NGRP] = acc[m][a][r] + bias;
        }
    }
}

// ---------------------------------------------------------------------------
// Kernel 1 (fallback, ws too small for W0 bf16): in-loop conversion,
// unroll-1 to keep static footprint small.
// ---------------------------------------------------------------------------
__global__ void __launch_bounds__(256)
k_group_gemm_cvt(const float* __restrict__ X, const float* __restrict__ W0,
                 const float* __restrict__ b0, float* __restrict__ GL)
{
    __shared__ unsigned sA[32 * (KC / 2)];

    const int tid   = threadIdx.x;
    const int wave  = tid >> 5;
    const int lane  = tid & 31;
    const int lh    = lane & 15;
    const bool hi   = (lane >= 16);
    const int rbase = blockIdx.y * 32;
    const int nbase = blockIdx.x * 512 + wave * 64;

    v8f acc[2][4];
    #pragma unroll
    for (int m = 0; m < 2; ++m)
        #pragma unroll
        for (int a = 0; a < 4; ++a)
            #pragma unroll
            for (int r = 0; r < 8; ++r) acc[m][a][r] = 0.0f;

    for (int k0 = 0; k0 < FDIM; k0 += KC) {
        __syncthreads();
        {
            const int row = tid >> 3;
            const int t8  = tid & 7;
            const float* src = X + (size_t)(rbase + row) * FDIM + k0;
            unsigned* dst = &sA[row * (KC / 2)];
            #pragma unroll 1
            for (int j = 0; j < (KC / 32); ++j) {
                const int c4 = t8 + j * 8;
                float4 v = ((const float4*)src)[c4];
                dst[c4 * 2]     = pk_bf16(v.x, v.y);
                dst[c4 * 2 + 1] = pk_bf16(v.z, v.w);
            }
        }
        __syncthreads();

        #pragma unroll 1
        for (int k = 0; k < KC; k += 32) {
            v8u afu[2];
            const int dbase = (k >> 1) + (hi ? 8 : 0);
            #pragma unroll
            for (int m = 0; m < 2; ++m) {
                const unsigned* p = &sA[(m * 16 + lh) * (KC / 2) + dbase];
                uint4 lo4 = *(const uint4*)(p);
                uint4 hi4 = *(const uint4*)(p + 4);
                afu[m][0] = lo4.x; afu[m][1] = lo4.y;
                afu[m][2] = lo4.z; afu[m][3] = lo4.w;
                afu[m][4] = hi4.x; afu[m][5] = hi4.y;
                afu[m][6] = hi4.z; afu[m][7] = hi4.w;
            }
            const v16bf afrag0 = __builtin_bit_cast(v16bf, afu[0]);
            const v16bf afrag1 = __builtin_bit_cast(v16bf, afu[1]);

            #pragma unroll 1
            for (int a = 0; a < 4; ++a) {
                const int col = nbase + a * 16 + lh;
                const float* bp =
                    W0 + (size_t)col * FDIM + (k0 + k) + (hi ? 16 : 0);
                float4 q0 = ((const float4*)bp)[0];
                float4 q1 = ((const float4*)bp)[1];
                float4 q2 = ((const float4*)bp)[2];
                float4 q3 = ((const float4*)bp)[3];
                v8u bu;
                bu[0] = pk_bf16(q0.x, q0.y); bu[1] = pk_bf16(q0.z, q0.w);
                bu[2] = pk_bf16(q1.x, q1.y); bu[3] = pk_bf16(q1.z, q1.w);
                bu[4] = pk_bf16(q2.x, q2.y); bu[5] = pk_bf16(q2.z, q2.w);
                bu[6] = pk_bf16(q3.x, q3.y); bu[7] = pk_bf16(q3.z, q3.w);
                const v16bf bfrag = __builtin_bit_cast(v16bf, bu);

                acc[0][a] = __builtin_amdgcn_wmma_f32_16x16x32_bf16(
                    false, afrag0, false, bfrag, (short)0, acc[0][a], false, false);
                acc[1][a] = __builtin_amdgcn_wmma_f32_16x16x32_bf16(
                    false, afrag1, false, bfrag, (short)0, acc[1][a], false, false);
            }
        }
    }

    #pragma unroll
    for (int a = 0; a < 4; ++a) {
        const int col  = nbase + a * 16 + lh;
        const float bias = b0[col];
        #pragma unroll
        for (int m = 0; m < 2; ++m) {
            const int row0 = rbase + m * 16 + (hi ? 8 : 0);
            float* out = GL + (size_t)row0 * NGRP + col;
            #pragma unroll
            for (int r = 0; r < 8; ++r)
                out[(size_t)r * NGRP] = acc[m][a][r] + bias;
        }
    }
}

// ---------------------------------------------------------------------------
// Kernel 2 (fast path): emb = X @ W1^T + b1  (bf16 operands from ws)
// One 16x16 wmma tile per wave.
// ---------------------------------------------------------------------------
__global__ void __launch_bounds__(256)
k_emb_gemm_bf(const unsigned* __restrict__ Xb, const unsigned* __restrict__ W1b,
              const float* __restrict__ b1, float* __restrict__ Emb)
{
    const int tid  = threadIdx.x;
    const int wave = tid >> 5;
    const int lane = tid & 31;
    const int lh   = lane & 15;
    const bool hi  = (lane >= 16);

    const int nt = blockIdx.x * 8 + wave;
    if (nt >= (HDIM + 15) / 16) return;
    const int mbase = blockIdx.y * 16;

    const int bcol = nt * 16 + lh;
    const int ccol = (bcol < HDIM) ? bcol : (HDIM - 1);

    v8f acc;
    #pragma unroll
    for (int r = 0; r < 8; ++r) acc[r] = 0.0f;

    const unsigned* arow = Xb  + (size_t)(mbase + lh) * LDW + (hi ? 8 : 0);
    const unsigned* brow = W1b + (size_t)ccol * LDW + (hi ? 8 : 0);

    for (int k = 0; k < FDIM; k += 32) {
        const unsigned* ap = arow + (k >> 1);
        uint4 a0 = *(const uint4*)(ap);
        uint4 a1 = *(const uint4*)(ap + 4);
        v8u au;
        au[0] = a0.x; au[1] = a0.y; au[2] = a0.z; au[3] = a0.w;
        au[4] = a1.x; au[5] = a1.y; au[6] = a1.z; au[7] = a1.w;

        const unsigned* bp = brow + (k >> 1);
        uint4 b0v = *(const uint4*)(bp);
        uint4 b1v = *(const uint4*)(bp + 4);
        v8u bu;
        bu[0] = b0v.x; bu[1] = b0v.y; bu[2] = b0v.z; bu[3] = b0v.w;
        bu[4] = b1v.x; bu[5] = b1v.y; bu[6] = b1v.z; bu[7] = b1v.w;

        acc = __builtin_amdgcn_wmma_f32_16x16x32_bf16(
            false, __builtin_bit_cast(v16bf, au),
            false, __builtin_bit_cast(v16bf, bu),
            (short)0, acc, false, false);
    }

    if (bcol < HDIM) {
        const float bias = b1[bcol];
        #pragma unroll
        for (int r = 0; r < 8; ++r) {
            const int row = mbase + r + (hi ? 8 : 0);
            Emb[(size_t)row * HDIM + bcol] = acc[r] + bias;
        }
    }
}

// ---------------------------------------------------------------------------
// Kernel 2 (fallback): in-loop conversion, unroll-1.
// ---------------------------------------------------------------------------
__global__ void __launch_bounds__(256)
k_emb_gemm_cvt(const float* __restrict__ X, const float* __restrict__ W1,
               const float* __restrict__ b1, float* __restrict__ Emb)
{
    const int tid  = threadIdx.x;
    const int wave = tid >> 5;
    const int lane = tid & 31;
    const int lh   = lane & 15;
    const bool hi  = (lane >= 16);

    const int nt = blockIdx.x * 8 + wave;
    if (nt >= (HDIM + 15) / 16) return;
    const int mbase = blockIdx.y * 16;

    const int bcol = nt * 16 + lh;
    const int ccol = (bcol < HDIM) ? bcol : (HDIM - 1);

    v8f acc;
    #pragma unroll
    for (int r = 0; r < 8; ++r) acc[r] = 0.0f;

    const float* arow = X  + (size_t)(mbase + lh) * FDIM;
    const float* brow = W1 + (size_t)ccol * FDIM;

    #pragma unroll 1
    for (int k = 0; k < FDIM; k += 32) {
        const float* ap = arow + k + (hi ? 16 : 0);
        float4 x0 = ((const float4*)ap)[0];
        float4 x1 = ((const float4*)ap)[1];
        float4 x2 = ((const float4*)(ap + 8))[0];
        float4 x3 = ((const float4*)(ap + 8))[1];
        v8u au;
        au[0] = pk_bf16(x0.x, x0.y); au[1] = pk_bf16(x0.z, x0.w);
        au[2] = pk_bf16(x1.x, x1.y); au[3] = pk_bf16(x1.z, x1.w);
        au[4] = pk_bf16(x2.x, x2.y); au[5] = pk_bf16(x2.z, x2.w);
        au[6] = pk_bf16(x3.x, x3.y); au[7] = pk_bf16(x3.z, x3.w);

        const float* bp = brow + k + (hi ? 16 : 0);
        float4 q0 = ((const float4*)bp)[0];
        float4 q1 = ((const float4*)bp)[1];
        float4 q2 = ((const float4*)bp)[2];
        float4 q3 = ((const float4*)bp)[3];
        v8u bu;
        bu[0] = pk_bf16(q0.x, q0.y); bu[1] = pk_bf16(q0.z, q0.w);
        bu[2] = pk_bf16(q1.x, q1.y); bu[3] = pk_bf16(q1.z, q1.w);
        bu[4] = pk_bf16(q2.x, q2.y); bu[5] = pk_bf16(q2.z, q2.w);
        bu[6] = pk_bf16(q3.x, q3.y); bu[7] = pk_bf16(q3.z, q3.w);

        acc = __builtin_amdgcn_wmma_f32_16x16x32_bf16(
            false, __builtin_bit_cast(v16bf, au),
            false, __builtin_bit_cast(v16bf, bu),
            (short)0, acc, false, false);
    }

    if (bcol < HDIM) {
        const float bias = b1[bcol];
        #pragma unroll
        for (int r = 0; r < 8; ++r) {
            const int row = mbase + r + (hi ? 8 : 0);
            Emb[(size_t)row * HDIM + bcol] = acc[r] + bias;
        }
    }
}

// ---------------------------------------------------------------------------
// Kernel 3: per-row top-10 of group_logits (sigmoid monotone -> topk on
// logits). One block per batch row, LDS copy, 10 argmax rounds, low-index
// tie-break (matches jax.lax.top_k).
// ---------------------------------------------------------------------------
__global__ void __launch_bounds__(256)
k_topk(const float* __restrict__ GL, int* __restrict__ top_idx,
       float* __restrict__ top_scores)
{
    __shared__ float sV[NGRP];
    __shared__ float rV[256];
    __shared__ int   rI[256];

    const int b   = blockIdx.x;
    const int tid = threadIdx.x;
    const float* row = GL + (size_t)b * NGRP;

    for (int i = tid; i < NGRP; i += 256) sV[i] = row[i];
    __syncthreads();

    for (int t = 0; t < TOPK; ++t) {
        float bv = -INFINITY;
        int   bi = 0x7FFFFFFF;
        for (int i = tid; i < NGRP; i += 256) {
            const float v = sV[i];
            if (v > bv || (v == bv && i < bi)) { bv = v; bi = i; }
        }
        rV[tid] = bv; rI[tid] = bi;
        __syncthreads();
        for (int s = 128; s > 0; s >>= 1) {
            if (tid < s) {
                const float ov = rV[tid + s]; const int oi = rI[tid + s];
                if (ov > rV[tid] || (ov == rV[tid] && oi < rI[tid])) {
                    rV[tid] = ov; rI[tid] = oi;
                }
            }
            __syncthreads();
        }
        if (tid == 0) {
            top_idx[b * TOPK + t]    = rI[0];
            top_scores[b * TOPK + t] = sigmoidf_(rV[0]);
            sV[rI[0]] = -INFINITY;
        }
        __syncthreads();
    }
}

// ---------------------------------------------------------------------------
// Kernel 4: candidates + embedding gather + per-sample dot + final scores.
// Memory-bound gather (252 MB ~ 11 us at HBM peak): wave32 per candidate.
// ---------------------------------------------------------------------------
__global__ void __launch_bounds__(256)
k_cand(const float* __restrict__ Emb, const float* __restrict__ embedT,
       const int* __restrict__ group_y, const int* __restrict__ top_idx,
       const float* __restrict__ top_scores,
       int* __restrict__ out_cand, float* __restrict__ out_scores)
{
    __shared__ __align__(16) float sE[HDIM + 4];

    const int b    = blockIdx.y;
    const int tid  = threadIdx.x;
    const int wave = tid >> 5;
    const int lane = tid & 31;

    for (int i = tid; i < HDIM; i += 256) sE[i] = Emb[(size_t)b * HDIM + i];
    __syncthreads();

    const int c = blockIdx.x * 8 + wave;
    if (c >= NCAND) return;

    const int kk = c / SLAB;
    const int s  = c - kk * SLAB;
    const int gi = top_idx[b * TOPK + kk];
    const int label = group_y[(size_t)gi * SLAB + s];

    const float4* er = (const float4*)(embedT + (size_t)label * HDIM);
    const float4* ev = (const float4*)sE;
    float sum = 0.0f;
    #pragma unroll
    for (int j = 0; j < 3; ++j) {
        const int i4 = lane + j * 32;
        if (i4 < HDIM / 4) {
            const float4 x = er[i4];
            const float4 y = ev[i4];
            sum += x.x * y.x + x.y * y.y + x.z * y.z + x.w * y.w;
        }
    }
    #pragma unroll
    for (int off = 16; off > 0; off >>= 1)
        sum += __shfl_xor(sum, off, 32);

    if (lane == 0) {
        out_cand[(size_t)b * NCAND + c]   = label;
        out_scores[(size_t)b * NCAND + c] =
            sigmoidf_(sum) * top_scores[b * TOPK + kk];
    }
}

// ---------------------------------------------------------------------------
// Workspace layout (bytes)
// ---------------------------------------------------------------------------
static constexpr size_t SZ_EMB  = (size_t)BATCH * HDIM * 4;     // 307200
static constexpr size_t SZ_TIDX = (size_t)BATCH * TOPK * 4;     // 10240
static constexpr size_t SZ_TSC  = (size_t)BATCH * TOPK * 4;     // 10240
static constexpr size_t OFF_XB  = ((SZ_EMB + SZ_TIDX + SZ_TSC + 255) / 256) * 256;
static constexpr size_t SZ_XB   = (size_t)BATCH * FDIM * 2;     // 1.97 MB
static constexpr size_t OFF_W1B = OFF_XB + SZ_XB;
static constexpr size_t SZ_W1B  = (size_t)HDIM * FDIM * 2;      // 2.30 MB
static constexpr size_t OFF_W0B = OFF_W1B + SZ_W1B;
static constexpr size_t SZ_W0B  = (size_t)NGRP * FDIM * 2;      // 62.9 MB

extern "C" void kernel_launch(void* const* d_in, const int* in_sizes, int n_in,
                              void* d_out, int out_size, void* d_ws, size_t ws_size,
                              hipStream_t stream)
{
    const float* X      = (const float*)d_in[0];   // [256,3840]
    const float* W0     = (const float*)d_in[1];   // [8192,3840]
    const float* b0     = (const float*)d_in[2];   // [8192]
    const float* W1     = (const float*)d_in[3];   // [300,3840]
    const float* b1     = (const float*)d_in[4];   // [300]
    const float* embedT = (const float*)d_in[5];   // [671744,300]
    const int*   gy     = (const int*)d_in[6];     // [8192,82]

    float* gl         = (float*)d_out;                          // [256,8192]
    int*   out_cand   = (int*)(gl + (size_t)BATCH * NGRP);      // [256,820]
    float* out_scores = gl + (size_t)BATCH * NGRP + (size_t)BATCH * NCAND;

    char*  ws         = (char*)d_ws;
    float* Emb        = (float*)ws;
    int*   top_idx    = (int*)(ws + SZ_EMB);
    float* top_scores = (float*)(ws + SZ_EMB + SZ_TIDX);
    unsigned* Xb      = (unsigned*)(ws + OFF_XB);
    unsigned* W1b     = (unsigned*)(ws + OFF_W1B);
    unsigned* W0b     = (unsigned*)(ws + OFF_W0B);

    const bool havePartial = ws_size >= OFF_W0B;            // X + W1 in bf16
    const bool haveFull    = ws_size >= OFF_W0B + SZ_W0B;   // + W0 in bf16

    if (havePartial) {
        k_cvt_bf16<<<dim3(512), 256, 0, stream>>>(X,  Xb,  BATCH * FDIM / 4);
        k_cvt_bf16<<<dim3(512), 256, 0, stream>>>(W1, W1b, HDIM * FDIM / 4);
        k_emb_gemm_bf<<<dim3(3, BATCH / 16), 256, 0, stream>>>(Xb, W1b, b1, Emb);
    } else {
        k_emb_gemm_cvt<<<dim3(3, BATCH / 16), 256, 0, stream>>>(X, W1, b1, Emb);
    }

    if (haveFull) {
        k_cvt_bf16<<<dim3(4096), 256, 0, stream>>>(W0, W0b, NGRP * FDIM / 4);
        k_group_gemm_bf<<<dim3(NGRP / 512, BATCH / 32), 256, 0, stream>>>(
            Xb, W0b, b0, gl);
    } else {
        k_group_gemm_cvt<<<dim3(NGRP / 512, BATCH / 32), 256, 0, stream>>>(
            X, W0, b0, gl);
    }

    k_topk<<<dim3(BATCH), 256, 0, stream>>>(gl, top_idx, top_scores);
    k_cand<<<dim3((NCAND + 7) / 8, BATCH), 256, 0, stream>>>(
        Emb, embedT, gy, top_idx, top_scores, out_cand, out_scores);
}
